// GHNN_20375324852682
// MI455X (gfx1250) — matrix operand order repaired and checked by reading, not compiled, not run on hardware
//
#include <hip/hip_runtime.h>
#include <hip/hip_bf16.h>
#include <math.h>

typedef __attribute__((ext_vector_type(16))) __bf16 v16bf;
typedef __attribute__((ext_vector_type(8)))  float  v8f;

#define ALPHA_C 0.1f
#define PPR_ITERS 10
#define NF 512
#define NH 64
#define NC 64
#define TROWS 64
#define KSTEP 32

// ---------------- WMMA helpers ----------------

__device__ __forceinline__ v8f wmma_bf16(v16bf a, v16bf b, v8f c) {
  // D = A*B + C, 16x16x32 bf16 -> f32
  return __builtin_amdgcn_wmma_f32_16x16x32_bf16(false, a, false, b, (short)0, c,
                                                 false, false);
}

__device__ __forceinline__ void bfsplit(float v, __bf16* hi, __bf16* lo) {
  __bf16 h = (__bf16)v;
  *hi = h;
  *lo = (__bf16)(v - (float)h);
}

// Fragment-major index helpers.
// A-matrix (16-bit 16x32): lane = (r&15) + 16*(k>=16),
//   element e = k&15  (k = K within the 32-wide step)
__device__ __forceinline__ int a_lane_hi(int k) { return ((k >> 4) & 1) << 4; }
__device__ __forceinline__ int a_elem(int k) { return k & 15; }
// B-matrix (16-bit Kx16, row striped across lanes): lane = (n&15) + 16*(k>=16), e = k&15

// ---------------- Fused MLP: z0 = h = relu(x@W1+b1)@W2+b2 ----------------
//
// 128 threads (4 waves), 64 output rows per block; each wave owns a 16x64 strip.
// All LDS tiles are stored in WMMA fragment-major layout so each fragment load
// is one aligned 32B vector load (2x ds_load_b128) per operand.

__global__ __launch_bounds__(128) void mlp_kernel(
    const float* __restrict__ x, const float* __restrict__ W1,
    const float* __restrict__ b1, const float* __restrict__ W2,
    const float* __restrict__ b2, float* __restrict__ h_out,
    float* __restrict__ z_out, int N) {
  // frag-major: [tile][lane] -> 16 contiguous bf16 (32B)
  __shared__ v16bf sAh[4 * 32],     sAl[4 * 32];      // A tile: 4 waves
  __shared__ v16bf sBh[4 * 32],     sBl[4 * 32];      // B tile: 4 n-tiles
  __shared__ v16bf sHh[2 * 4 * 32], sHl[2 * 4 * 32];  // h tile: 2 k-steps x 4 waves
  __shared__ v16bf sW2h[2 * 4 * 32], sW2l[2 * 4 * 32]; // W2: 2 k-steps x 4 n-tiles

  const int tid  = threadIdx.x;
  const int wave = tid >> 5;
  const int lane = tid & 31;
  const int row0 = blockIdx.x * TROWS;

  // Preload W2 (64x64) into B-fragment layout, split hi/lo (once per block).
  {
    __bf16* wh = (__bf16*)sW2h;
    __bf16* wl = (__bf16*)sW2l;
    for (int i = tid; i < (NH * NC) / 4; i += 128) {
      float4 v = ((const float4*)W2)[i];
      int k = i >> 4, c = (i & 15) << 2;       // k: 0..63, c: 0..60 step 4
      int kt = k >> 5, kk = k & 31;
      int lhi = (kk >= 16) ? 16 : 0;
      int e = kk & 15;
      int nt = c >> 4, nl = c & 15;            // 4 consecutive cols, same 16-group
      int idx = (((kt * 4 + nt) * 32 + lhi + nl) << 4) + e;
      bfsplit(v.x, wh + idx,      wl + idx);
      bfsplit(v.y, wh + idx + 16, wl + idx + 16);
      bfsplit(v.z, wh + idx + 32, wl + idx + 32);
      bfsplit(v.w, wh + idx + 48, wl + idx + 48);
    }
  }

  const v8f vzero = {0.f, 0.f, 0.f, 0.f, 0.f, 0.f, 0.f, 0.f};
  v8f acc[4];
#pragma unroll
  for (int t = 0; t < 4; ++t) acc[t] = vzero;

  // ---- GEMM1: [64 x 512] @ [512 x 64], K-steps of 32, split-bf16 (3 WMMA) ----
  for (int kt = 0; kt < NF / KSTEP; ++kt) {
    // stage A tile 64x32 into A-fragment layout (hi/lo)
    {
      __bf16* ah = (__bf16*)sAh;
      __bf16* al = (__bf16*)sAl;
      for (int i = tid; i < (TROWS * KSTEP) / 4; i += 128) {
        int r = i >> 3, c = (i & 7) << 2;      // r: 0..63, c: 0..28 step 4
        int gr = row0 + r;
        float4 v = (gr < N)
            ? *(const float4*)(x + (size_t)gr * NF + kt * KSTEP + c)
            : make_float4(0.f, 0.f, 0.f, 0.f);
        int w = r >> 4, mm = r & 15;
        int idx = (((w * 32) + a_lane_hi(c) + mm) << 4) + a_elem(c);
        bfsplit(v.x, ah + idx,     al + idx);
        bfsplit(v.y, ah + idx + 1, al + idx + 1);
        bfsplit(v.z, ah + idx + 2, al + idx + 2);
        bfsplit(v.w, ah + idx + 3, al + idx + 3);
      }
    }
    // stage B tile 32x64 from W1 into B-fragment layout (hi/lo)
    {
      __bf16* bh = (__bf16*)sBh;
      __bf16* bl = (__bf16*)sBl;
      for (int i = tid; i < (KSTEP * NH) / 4; i += 128) {
        int k = i >> 4, c = (i & 15) << 2;     // k: 0..31, c: 0..60 step 4
        float4 v = *(const float4*)(W1 + (size_t)(kt * KSTEP + k) * NH + c);
        int lhi = (k >= 16) ? 16 : 0;
        int e = k & 15;
        int nt = c >> 4, nl = c & 15;
        int idx = (((nt * 32) + lhi + nl) << 4) + e;
        bfsplit(v.x, bh + idx,      bl + idx);
        bfsplit(v.y, bh + idx + 16, bl + idx + 16);
        bfsplit(v.z, bh + idx + 32, bl + idx + 32);
        bfsplit(v.w, bh + idx + 48, bl + idx + 48);
      }
    }
    __syncthreads();

    v16bf ah = sAh[wave * 32 + lane];
    v16bf al = sAl[wave * 32 + lane];
#pragma unroll
    for (int nt = 0; nt < 4; ++nt) {
      v16bf bh = sBh[nt * 32 + lane];
      v16bf bl = sBl[nt * 32 + lane];
      acc[nt] = wmma_bf16(ah, bh, acc[nt]);
      acc[nt] = wmma_bf16(ah, bl, acc[nt]);
      acc[nt] = wmma_bf16(al, bh, acc[nt]);
    }
    __syncthreads();
  }

  // bias + ReLU; stage h tile (64x64) into A-fragment layout for GEMM2
  {
    __bf16* hh = (__bf16*)sHh;
    __bf16* hl = (__bf16*)sHl;
#pragma unroll
    for (int nt = 0; nt < 4; ++nt) {
      int n = nt * 16 + (lane & 15);   // this is GEMM2's K coordinate
      int kt = n >> 5, kk = n & 31;
      int lhi = a_lane_hi(kk);
      int e = a_elem(kk);
      float bias = b1[n];
#pragma unroll
      for (int r = 0; r < 8; ++r) {
        float v = fmaxf(acc[nt][r] + bias, 0.f);
        int rl = wave * 16 + r + ((lane >> 4) << 3);  // C layout: lanes>=16 -> M=r+8
        int idx = ((((kt * 4 + wave) * 32) + lhi + (rl & 15)) << 4) + e;
        bfsplit(v, hh + idx, hl + idx);
      }
    }
  }
  __syncthreads();

  // ---- GEMM2: [64 x 64] @ [64 x 64], 2 K-steps, split-bf16 ----
  v8f acc2[4];
#pragma unroll
  for (int t = 0; t < 4; ++t) acc2[t] = vzero;

#pragma unroll
  for (int kt = 0; kt < 2; ++kt) {
    v16bf ah = sHh[(kt * 4 + wave) * 32 + lane];
    v16bf al = sHl[(kt * 4 + wave) * 32 + lane];
#pragma unroll
    for (int nt = 0; nt < 4; ++nt) {
      v16bf bh = sW2h[(kt * 4 + nt) * 32 + lane];
      v16bf bl = sW2l[(kt * 4 + nt) * 32 + lane];
      acc2[nt] = wmma_bf16(ah, bh, acc2[nt]);
      acc2[nt] = wmma_bf16(ah, bl, acc2[nt]);
      acc2[nt] = wmma_bf16(al, bh, acc2[nt]);
    }
  }

  // bias + store h and z0
#pragma unroll
  for (int nt = 0; nt < 4; ++nt) {
    int n = nt * 16 + (lane & 15);
    float bias = b2[n];
#pragma unroll
    for (int r = 0; r < 8; ++r) {
      int gr = row0 + wave * 16 + r + ((lane >> 4) << 3);
      if (gr < N) {
        float v = acc2[nt][r] + bias;
        h_out[(size_t)gr * NC + n] = v;
        z_out[(size_t)gr * NC + n] = v;
      }
    }
  }
}

// ---------------- Degree / normalization ----------------

__global__ void deg_init_kernel(float* __restrict__ deg, int N) {
  int i = blockIdx.x * blockDim.x + threadIdx.x;
  if (i < N) deg[i] = 1.0f;  // self-loop
}

__global__ void deg_acc_kernel(const int* __restrict__ row, float* __restrict__ deg, int E) {
  int e = blockIdx.x * blockDim.x + threadIdx.x;
  if (e < E) unsafeAtomicAdd(&deg[row[e]], 1.0f);
}

__global__ void dinv_kernel(const float* __restrict__ deg, float* __restrict__ dinv, int N) {
  int i = blockIdx.x * blockDim.x + threadIdx.x;
  if (i < N) dinv[i] = rsqrtf(fmaxf(deg[i], 1.0f));
}

// ---------------- Propagation ----------------

// z_next = alpha*h + (1-alpha)*dinv[i]^2*z   (self-loop folded in, dense init)
__global__ void prop_init_kernel(const float* __restrict__ h, const float* __restrict__ z,
                                 const float* __restrict__ dinv,
                                 float* __restrict__ znext, int total) {
  int idx = blockIdx.x * blockDim.x + threadIdx.x;
  if (idx < total) {
    float di = dinv[idx >> 6];
    znext[idx] = ALPHA_C * h[idx] + (1.0f - ALPHA_C) * di * di * z[idx];
  }
}

// one wave32 per edge; each lane handles 2 of the 64 channels
__global__ __launch_bounds__(256) void prop_edge_kernel(
    const int* __restrict__ row, const int* __restrict__ col,
    const float* __restrict__ dinv, const float* __restrict__ z,
    float* __restrict__ znext, int E) {
  int e = blockIdx.x * 8 + (threadIdx.x >> 5);
  if (e >= E) return;
  int lane = threadIdx.x & 31;
  int r = row[e];
  int c = col[e];
  float w = (1.0f - ALPHA_C) * dinv[r] * dinv[c];
  float2 v = *(const float2*)(z + (size_t)c * NC + lane * 2);
  float* dst = znext + (size_t)r * NC + lane * 2;
  unsafeAtomicAdd(dst + 0, w * v.x);
  unsafeAtomicAdd(dst + 1, w * v.y);
}

// ---------------- Log-softmax (one wave per node) ----------------

__global__ __launch_bounds__(256) void lsm_kernel(const float* __restrict__ z,
                                                  float* __restrict__ out, int N) {
  int i = blockIdx.x * 8 + (threadIdx.x >> 5);
  if (i >= N) return;
  int lane = threadIdx.x & 31;
  float2 v = *(const float2*)(z + (size_t)i * NC + lane * 2);
  float m = fmaxf(v.x, v.y);
#pragma unroll
  for (int off = 16; off > 0; off >>= 1) m = fmaxf(m, __shfl_xor(m, off, 32));
  float s = expf(v.x - m) + expf(v.y - m);
#pragma unroll
  for (int off = 16; off > 0; off >>= 1) s += __shfl_xor(s, off, 32);
  float lse = m + logf(s);
  float2 o;
  o.x = v.x - lse;
  o.y = v.y - lse;
  *(float2*)(out + (size_t)i * NC + lane * 2) = o;
}

// ---------------- Launch ----------------

extern "C" void kernel_launch(void* const* d_in, const int* in_sizes, int n_in,
                              void* d_out, int out_size, void* d_ws, size_t ws_size,
                              hipStream_t stream) {
  const float* x    = (const float*)d_in[0];
  const int*   edge = (const int*)d_in[1];   // [2, E] int32
  const float* W1   = (const float*)d_in[2];
  const float* b1   = (const float*)d_in[3];
  const float* W2   = (const float*)d_in[4];
  const float* b2   = (const float*)d_in[5];
  float* out = (float*)d_out;

  const int N = in_sizes[0] / NF;
  const int E = in_sizes[1] / 2;

  float* h    = (float*)d_ws;
  float* zA   = h + (size_t)N * NC;
  float* zB   = zA + (size_t)N * NC;
  float* deg  = zB + (size_t)N * NC;
  float* dinv = deg + N;

  // fused MLP -> h (and z0 = h)
  mlp_kernel<<<(N + TROWS - 1) / TROWS, 128, 0, stream>>>(x, W1, b1, W2, b2, h, zA, N);

  // degrees (targets) + self loops, then D^-1/2
  deg_init_kernel<<<(N + 255) / 256, 256, 0, stream>>>(deg, N);
  deg_acc_kernel<<<(E + 255) / 256, 256, 0, stream>>>(edge, deg, E);
  dinv_kernel<<<(N + 255) / 256, 256, 0, stream>>>(deg, dinv, N);

  // 10 PPR iterations, ping-pong z buffers
  float* zin = zA;
  float* znx = zB;
  const int total = N * NC;
  for (int it = 0; it < PPR_ITERS; ++it) {
    prop_init_kernel<<<(total + 255) / 256, 256, 0, stream>>>(h, zin, dinv, znx, total);
    prop_edge_kernel<<<(E + 7) / 8, 256, 0, stream>>>(edge, edge + E, dinv, zin, znx, E);
    float* t = zin; zin = znx; znx = t;
  }

  // log-softmax over 64 classes per node
  lsm_kernel<<<(N + 7) / 8, 256, 0, stream>>>(zin, out, N);
}
